// LinFormerSelfAttention_74371653698142
// MI455X (gfx1250) — compile-verified, matrix operand order checked by reading, not run on hardware
//
#include <hip/hip_runtime.h>

// ---------------------------------------------------------------------------
// Linformer self-attention for MI455X (gfx1250), wave32 + WMMA f16.
// All matmuls via v_wmma_f32_16x16x32_f16; f32 accumulate; f32 softmax.
// Contiguous tile staging uses global_load_async_to_lds_b128 (ASYNCcnt);
// transpose staging stays VGPR->ds_store. GEMM is double-buffered.
// N padded 870 -> 896 (multiple of 128) so every tile/K-loop is exact.
// ---------------------------------------------------------------------------

typedef _Float16 half_t;
typedef __attribute__((ext_vector_type(16))) _Float16 v16h;
typedef __attribute__((ext_vector_type(8)))  float    v8f;

#define BB   64
#define NN   870
#define NP   896
#define DD   512
#define HH   2
#define DH   256
#define KP   256
#define TLD  40      // LDS tile leading dim (half_t): 80B rows -> 16B aligned, 20-dword
                     // stride is bank-conflict free (20*m mod 64 distinct for m=0..15)

union FragU { v16h v; unsigned int u[8]; };

// Async copy 16 bytes global -> LDS (tracked on ASYNCcnt). LDS VGPR operand is
// the ds-relative byte address == low 32 bits of the generic pointer.
__device__ __forceinline__ void async_b128(const half_t* g, half_t* l) {
  unsigned loff = (unsigned)(unsigned long long)(void*)l;
  asm volatile("global_load_async_to_lds_b128 %0, %1, off"
               :: "v"(loff), "v"(g)
               : "memory");
}
__device__ __forceinline__ void wait_async0() {
  asm volatile("s_wait_asynccnt 0x0" ::: "memory");
}

// Load a 16x32 f16 fragment (A-matrix layout per CDNA5 ISA 7.12.2) from a
// row-major tile (rows = M for A, rows = N for transposed-staged B).
__device__ __forceinline__ v16h load_frag16x32(const half_t* tile, int ld) {
  const int lane = (int)(threadIdx.x & 31);
  const int m  = lane & 15;
  const int hl = lane >> 4;
  FragU f;
#pragma unroll
  for (int r = 0; r < 8; ++r) {
    const int kb = ((r & 4) << 2) + (hl << 3) + ((r & 3) << 1); // K base: {0|16} + hl*8 + 2*(r&3)
    f.u[r] = *reinterpret_cast<const unsigned int*>(tile + m * ld + kb);
  }
  return f.v;
}

// ---------------------------------------------------------------------------
// Prep kernels
// ---------------------------------------------------------------------------
__global__ __launch_bounds__(256) void prep_x(const float* __restrict__ x,
                                              const unsigned char* __restrict__ mask,
                                              half_t* __restrict__ xb) {
  long long i = (long long)blockIdx.x * 256 + threadIdx.x;   // < BB*NP*DD
  int d = (int)(i & (DD - 1));
  long long r = i >> 9;                                      // b*NP + n
  int n = (int)(r % NP);
  int b = (int)(r / NP);
  float v = 0.f;
  if (n < NN) {
    if (!mask[b * NN + n]) v = x[((long long)b * NN + n) * DD + d];
  }
  xb[i] = (half_t)v;
}

__global__ __launch_bounds__(256) void conv_f32_f16(const float* __restrict__ s,
                                                    half_t* __restrict__ d, int n) {
  int i = blockIdx.x * 256 + threadIdx.x;
  if (i < n) d[i] = (half_t)s[i];
}

// proj [NN, KP] f32  ->  projT [KP, NP] f16 (zero-padded columns NN..NP-1)
__global__ __launch_bounds__(256) void prep_projT(const float* __restrict__ p,
                                                  half_t* __restrict__ pT) {
  int i = blockIdx.x * 256 + threadIdx.x;   // < KP*NP
  int n = i % NP;
  int k = i / NP;
  float v = (n < NN) ? p[(long long)n * KP + k] : 0.f;
  pT[i] = (half_t)v;
}

// ---------------------------------------------------------------------------
// Generic batched f16 GEMM: C[b] = A[b] (MxK, row-major) * B[b] (KxN, row-major)
// Block tile 128x128, 256 threads = 8 waves, wave tile 32x64 (2x4 WMMA tiles).
// Double-buffered K loop: A staged by async-to-LDS b128, B staged transposed
// via VGPR scatter. One barrier per iteration. grid = (N/128, ceil(M/128), b).
// ---------------------------------------------------------------------------
template <bool OUT_F32>
__global__ __launch_bounds__(256) void gemm_f16(const half_t* __restrict__ A,
                                                const half_t* __restrict__ Bm,
                                                void* __restrict__ Cv,
                                                int M, int Kdim,
                                                int lda, int ldb, int ldc,
                                                long long sAb, long long sBb, long long sCb) {
  __shared__ half_t sA[2][128][TLD];
  __shared__ half_t sB[2][128][TLD];   // B tile stored transposed: [n][k]

  const int tid  = (int)threadIdx.x;
  const int b    = (int)blockIdx.z;
  const int lane = tid & 31;
  const int w    = tid >> 5;
  const int wm   = w >> 1;   // 0..3 -> M offset wm*32
  const int wn   = w & 1;    // 0..1 -> N offset wn*64

  const half_t* Ab = A  + (long long)b * sAb + (long long)blockIdx.y * 128 * lda;
  const half_t* Bb = Bm + (long long)b * sBb + (long long)blockIdx.x * 128;

  auto stageA = [&](int buf, int k0) {           // 128x32 halves = 512 b128 chunks
#pragma unroll
    for (int i = 0; i < 2; ++i) {
      int li = tid + i * 256;                    // 0..511
      int r  = li >> 2;                          // 0..127
      int c  = (li & 3) << 3;                    // 0,8,16,24
      async_b128(Ab + (long long)r * lda + k0 + c, &sA[buf][r][c]);
    }
  };
  auto stageB = [&](int buf, int k0) {           // transpose scatter
#pragma unroll
    for (int i = 0; i < 8; ++i) {
      int li = tid + i * 256;                    // 0..2047
      int kr = li >> 6;                          // 0..31
      int nc = (li & 63) << 1;                   // 0..126
      unsigned int dw =
          *reinterpret_cast<const unsigned int*>(Bb + (long long)(k0 + kr) * ldb + nc);
      sB[buf][nc][kr]     = __builtin_bit_cast(half_t, (unsigned short)(dw & 0xffffu));
      sB[buf][nc + 1][kr] = __builtin_bit_cast(half_t, (unsigned short)(dw >> 16));
    }
  };

  v8f acc[2][4];
#pragma unroll
  for (int i = 0; i < 2; ++i)
#pragma unroll
    for (int j = 0; j < 4; ++j)
#pragma unroll
      for (int e = 0; e < 8; ++e) acc[i][j][e] = 0.f;

  const int nk = Kdim >> 5;
  stageA(0, 0);
  stageB(0, 0);
  wait_async0();
  __syncthreads();

  for (int ki = 0; ki < nk; ++ki) {
    const int cur = ki & 1;
    if (ki + 1 < nk) {                 // prefetch next K-step into the other buffer
      stageA(cur ^ 1, (ki + 1) << 5);
      stageB(cur ^ 1, (ki + 1) << 5);
    }
    v16h aF[2], bF[4];
#pragma unroll
    for (int tm = 0; tm < 2; ++tm)
      aF[tm] = load_frag16x32(&sA[cur][wm * 32 + tm * 16][0], TLD);
#pragma unroll
    for (int tn = 0; tn < 4; ++tn)
      bF[tn] = load_frag16x32(&sB[cur][wn * 64 + tn * 16][0], TLD);
#pragma unroll
    for (int tm = 0; tm < 2; ++tm)
#pragma unroll
      for (int tn = 0; tn < 4; ++tn)
        acc[tm][tn] = __builtin_amdgcn_wmma_f32_16x16x32_f16(
            false, aF[tm], false, bF[tn], (short)0, acc[tm][tn], false, false);
    wait_async0();
    __syncthreads();
  }

  // Store: C/D layout -> VGPR j holds row j (lanes 0-15) / j+8 (lanes 16-31), col = lane&15.
  const int nl    = lane & 15;
  const int rbase = (lane >> 4) * 8;
  const long long row0 = (long long)blockIdx.y * 128 + wm * 32;
  const long long col0 = (long long)blockIdx.x * 128 + wn * 64;
#pragma unroll
  for (int tm = 0; tm < 2; ++tm)
#pragma unroll
    for (int tn = 0; tn < 4; ++tn)
#pragma unroll
      for (int j = 0; j < 8; ++j) {
        long long row = row0 + tm * 16 + rbase + j;
        if (row < M) {
          long long col = col0 + tn * 16 + nl;
          if (OUT_F32) {
            float* C = (float*)Cv + (long long)b * sCb;
            C[row * ldc + col] = acc[tm][tn][j];
          } else {
            half_t* C = (half_t*)Cv + (long long)b * sCb;
            C[row * ldc + col] = (half_t)acc[tm][tn][j];
          }
        }
      }
}

// ---------------------------------------------------------------------------
// Fused attention: per (b, h, 64-row tile of N):
//   dots = q_tile[64,DH] * keys[b,h]^T[DH,KP] * scale   (WMMA, f32 acc in LDS)
//   attn = softmax(dots, axis=KP)                       (f32, -> f16 in LDS)
//   out  = attn[64,KP] * vals[b,h][KP,DH]               (WMMA -> f16 global)
// Phase-1 staging uses async-to-LDS b128 (contiguous rows); phase-2 B needs a
// transpose so it stays a VGPR scatter.
// ---------------------------------------------------------------------------
__global__ __launch_bounds__(256) void attn_kernel(const half_t* __restrict__ q,
                                                   const half_t* __restrict__ keys,
                                                   const half_t* __restrict__ vals,
                                                   half_t* __restrict__ out) {
  __shared__ float  sDots[64][260];     // 66.5 KB
  __shared__ half_t sAttn[64][258];     // 33.0 KB (even ld for dword frag loads)
  __shared__ half_t sT[256][TLD];       // 20.5 KB, B staging (both phases)
  __shared__ half_t sQ[64][TLD];        //  5.1 KB

  const int tid  = (int)threadIdx.x;
  const int nt   = (int)blockIdx.x;     // 0..13
  const int h    = (int)blockIdx.y;     // 0..1
  const int b    = (int)blockIdx.z;
  const int hoff = h * DH;
  const long long qbase = ((long long)b * NP + nt * 64) * DD + hoff;
  const long long kbase = (long long)b * KP * DD + hoff;

  const int lane = tid & 31;
  const int w    = tid >> 5;
  const int wm   = w >> 2;   // 0..1 -> M offset wm*32
  const int wn   = w & 3;    // 0..3 -> N offset wn*64
  const int nl    = lane & 15;
  const int rbase = (lane >> 4) * 8;

  v8f acc[2][4];
#pragma unroll
  for (int i = 0; i < 2; ++i)
#pragma unroll
    for (int j = 0; j < 4; ++j)
#pragma unroll
      for (int e = 0; e < 8; ++e) acc[i][j][e] = 0.f;

  // ---- Phase 1: dots = q * keys^T  (inner dim = DH; keys rows are already the
  //      transposed-B layout we want in LDS, so rows copy contiguously)
  for (int d0 = 0; d0 < DH; d0 += 32) {
    __syncthreads();
    {                                            // sQ: 64 rows x 4 b128 chunks
      int r = tid >> 2;
      int c = (tid & 3) << 3;
      async_b128(q + qbase + (long long)r * DD + d0 + c, &sQ[r][c]);
    }
#pragma unroll
    for (int i = 0; i < 4; ++i) {                // sT: 256 rows x 4 b128 chunks
      int li = tid + i * 256;                    // 0..1023
      int kk = li >> 2;                          // 0..255
      int c  = (li & 3) << 3;
      async_b128(keys + kbase + (long long)kk * DD + d0 + c, &sT[kk][c]);
    }
    wait_async0();
    __syncthreads();
    v16h aF[2], bF[4];
#pragma unroll
    for (int tm = 0; tm < 2; ++tm) aF[tm] = load_frag16x32(&sQ[wm * 32 + tm * 16][0], TLD);
#pragma unroll
    for (int tn = 0; tn < 4; ++tn) bF[tn] = load_frag16x32(&sT[wn * 64 + tn * 16][0], TLD);
#pragma unroll
    for (int tm = 0; tm < 2; ++tm)
#pragma unroll
      for (int tn = 0; tn < 4; ++tn)
        acc[tm][tn] = __builtin_amdgcn_wmma_f32_16x16x32_f16(
            false, aF[tm], false, bF[tn], (short)0, acc[tm][tn], false, false);
  }
  __syncthreads();

  const float scale = 0.0625f;   // DH^-0.5 = 1/16
#pragma unroll
  for (int tm = 0; tm < 2; ++tm)
#pragma unroll
    for (int tn = 0; tn < 4; ++tn)
#pragma unroll
      for (int j = 0; j < 8; ++j)
        sDots[wm * 32 + tm * 16 + rbase + j][wn * 64 + tn * 16 + nl] = acc[tm][tn][j] * scale;
  __syncthreads();

  // ---- Softmax over KP=256, one row per thread (threads 0..63)
  if (tid < 64) {
    float mx = -1e30f;
    for (int kk = 0; kk < KP; ++kk) mx = fmaxf(mx, sDots[tid][kk]);
    float sum = 0.f;
    for (int kk = 0; kk < KP; ++kk) {
      float e = __expf(sDots[tid][kk] - mx);
      sDots[tid][kk] = e;
      sum += e;
    }
    float inv = 1.f / sum;
    for (int kk = 0; kk < KP; ++kk) sAttn[tid][kk] = (half_t)(sDots[tid][kk] * inv);
  }
  __syncthreads();

  // ---- Phase 2: out = attn * vals[b,h]  (inner dim = KP)
#pragma unroll
  for (int i = 0; i < 2; ++i)
#pragma unroll
    for (int j = 0; j < 4; ++j)
#pragma unroll
      for (int e = 0; e < 8; ++e) acc[i][j][e] = 0.f;

  for (int kk0 = 0; kk0 < KP; kk0 += 32) {
    __syncthreads();
    // sT[d][kc] = vals[b][kk0+kc][hoff+d]  (transposed stage, coalesced along d)
#pragma unroll
    for (int i = 0; i < 16; ++i) {
      int li = tid + i * 256;                  // 0..4095
      int kc = li >> 7;                        // 0..31
      int np2 = (li & 127) << 1;               // 0..254
      unsigned int dw = *reinterpret_cast<const unsigned int*>(
          vals + kbase + (long long)(kk0 + kc) * DD + np2);
      sT[np2][kc]     = __builtin_bit_cast(half_t, (unsigned short)(dw & 0xffffu));
      sT[np2 + 1][kc] = __builtin_bit_cast(half_t, (unsigned short)(dw >> 16));
    }
    __syncthreads();
    v16h aF[2], bF[4];
#pragma unroll
    for (int tm = 0; tm < 2; ++tm) aF[tm] = load_frag16x32(&sAttn[wm * 32 + tm * 16][kk0], 258);
#pragma unroll
    for (int tn = 0; tn < 4; ++tn) bF[tn] = load_frag16x32(&sT[wn * 64 + tn * 16][0], TLD);
#pragma unroll
    for (int tm = 0; tm < 2; ++tm)
#pragma unroll
      for (int tn = 0; tn < 4; ++tn)
        acc[tm][tn] = __builtin_amdgcn_wmma_f32_16x16x32_f16(
            false, aF[tm], false, bF[tn], (short)0, acc[tm][tn], false, false);
  }

  // Store out[b][nt*64+row][hoff+col] (padded rows carry garbage; final GEMM guards them)
#pragma unroll
  for (int tm = 0; tm < 2; ++tm)
#pragma unroll
    for (int tn = 0; tn < 4; ++tn)
#pragma unroll
      for (int j = 0; j < 8; ++j) {
        int row = wm * 32 + tm * 16 + rbase + j;
        int col = wn * 64 + tn * 16 + nl;
        out[((long long)b * NP + nt * 64 + row) * DD + hoff + col] = (half_t)acc[tm][tn][j];
      }
}

// ---------------------------------------------------------------------------
// Host launcher
// ---------------------------------------------------------------------------
extern "C" void kernel_launch(void* const* d_in, const int* in_sizes, int n_in,
                              void* d_out, int out_size, void* d_ws, size_t ws_size,
                              hipStream_t stream) {
  (void)in_sizes; (void)n_in; (void)out_size; (void)ws_size;
  const float*         x    = (const float*)d_in[0];
  const unsigned char* mask = (const unsigned char*)d_in[1];   // jax bool -> 1 byte
  const float*         Wq   = (const float*)d_in[2];
  const float*         Wk   = (const float*)d_in[3];
  const float*         Wv   = (const float*)d_in[4];
  const float*         pk   = (const float*)d_in[5];
  const float*         pv   = (const float*)d_in[6];
  const float*         Wo   = (const float*)d_in[7];
  float*               out  = (float*)d_out;

  size_t off = 0;
  auto wsAlloc = [&](size_t bytes) -> void* {
    void* p = (char*)d_ws + off;
    off += (bytes + 255) & ~(size_t)255;
    return p;
  };
  half_t* Wqb  = (half_t*)wsAlloc((size_t)DD * DD * 2);
  half_t* Wkb  = (half_t*)wsAlloc((size_t)DD * DD * 2);
  half_t* Wvb  = (half_t*)wsAlloc((size_t)DD * DD * 2);
  half_t* Wob  = (half_t*)wsAlloc((size_t)DD * DD * 2);
  half_t* pkT  = (half_t*)wsAlloc((size_t)KP * NP * 2);
  half_t* pvT  = (half_t*)wsAlloc((size_t)KP * NP * 2);
  half_t* xb   = (half_t*)wsAlloc((size_t)BB * NP * DD * 2);
  half_t* qb   = (half_t*)wsAlloc((size_t)BB * NP * DD * 2);
  half_t* tmp  = (half_t*)wsAlloc((size_t)BB * NP * DD * 2);  // kx -> vx -> attn-out
  half_t* keys = (half_t*)wsAlloc((size_t)BB * KP * DD * 2);
  half_t* vals = (half_t*)wsAlloc((size_t)BB * KP * DD * 2);

  // Prep
  prep_x<<<(BB * NP * DD) / 256, 256, 0, stream>>>(x, mask, xb);
  conv_f32_f16<<<(DD * DD) / 256, 256, 0, stream>>>(Wq, Wqb, DD * DD);
  conv_f32_f16<<<(DD * DD) / 256, 256, 0, stream>>>(Wk, Wkb, DD * DD);
  conv_f32_f16<<<(DD * DD) / 256, 256, 0, stream>>>(Wv, Wvb, DD * DD);
  conv_f32_f16<<<(DD * DD) / 256, 256, 0, stream>>>(Wo, Wob, DD * DD);
  prep_projT<<<(KP * NP) / 256, 256, 0, stream>>>(pk, pkT);
  prep_projT<<<(KP * NP) / 256, 256, 0, stream>>>(pv, pvT);

  const long long sX = (long long)NP * DD;   // per-batch stride of [NP,DD] tensors
  const long long sK = (long long)KP * DD;   // per-batch stride of [KP,DD] tensors
  dim3 gXW(DD / 128, NP / 128, BB);          // (4, 7, 64)
  dim3 gPR(DD / 128, KP / 128, BB);          // (4, 2, 64)

  // q = xb @ Wq
  gemm_f16<false><<<gXW, 256, 0, stream>>>(xb, Wqb, qb, NP, DD, DD, DD, DD, sX, 0, sX);
  // keys = pkT @ (xb @ Wk)
  gemm_f16<false><<<gXW, 256, 0, stream>>>(xb, Wkb, tmp, NP, DD, DD, DD, DD, sX, 0, sX);
  gemm_f16<false><<<gPR, 256, 0, stream>>>(pkT, tmp, keys, KP, NP, NP, DD, DD, 0, sX, sK);
  // vals = pvT @ (xb @ Wv)
  gemm_f16<false><<<gXW, 256, 0, stream>>>(xb, Wvb, tmp, NP, DD, DD, DD, DD, sX, 0, sX);
  gemm_f16<false><<<gPR, 256, 0, stream>>>(pvT, tmp, vals, KP, NP, NP, DD, DD, 0, sX, sK);
  // fused attention -> tmp (reused as attn output)
  attn_kernel<<<dim3(NP / 64, HH, BB), 256, 0, stream>>>(qb, keys, vals, tmp);
  // out = attn_out @ Wout (f32, guarded to N=870 rows)
  gemm_f16<true><<<gXW, 256, 0, stream>>>(tmp, Wob, out, NN, DD, DD, DD, DD, sX, 0,
                                          (long long)NN * DD);
}
